// Seq2SeqLSTM_2482491097416
// MI455X (gfx1250) — matrix operand order repaired and checked by reading, not compiled, not run on hardware
//
#include <hip/hip_runtime.h>
#include <hip/hip_bf16.h>

// ---------------------------------------------------------------------------
// Seq2Seq LSTM for MI455X (gfx1250, wave32, WMMA bf16 16x16x32 f32-acc)
//
// B=512, T_enc=256, T_dec=128, H=1024, gates=4H=4096, I=O=128.
// 32 workgroups x 16-row batch tiles; each WG runs the recurrence privately.
// 512 threads (16 waves, 4/SIMD -> ~256 VGPR/wave). Each wave owns 64 gate
// columns: 2 passes x (2 j-tiles x 4 gates) = 8 WMMA accumulator chains.
// Cell state c lives in per-thread LDS slots; K loops are rolled and
// sched-fenced; each k-iteration = 8-fragment load phase + WMMA phase so
// B loads clause together with graduated waits; next k-step warmed with
// WGP-scope global_prefetch_b8 (all cache levels, register-free).
// ---------------------------------------------------------------------------

typedef __attribute__((ext_vector_type(16))) __bf16 v16bf;
typedef __attribute__((ext_vector_type(8)))  __bf16 v8bf;
typedef __attribute__((ext_vector_type(8)))  float  v8f;

#define HID   1024
#define BATCH 512
#define TENC  256
#define TDEC  128
#define IND   128
#define OUTD  128
#define BT    16        // batch rows per workgroup
#define HP    1040      // hbuf pitch (bf16, 16-elem aligned + pad)
#define XP    144       // xbuf pitch
#define NTHR  512

#if __has_builtin(__builtin_amdgcn_sched_barrier)
#define SCHED_FENCE() __builtin_amdgcn_sched_barrier(0)
#else
#define SCHED_FENCE() asm volatile("" ::: "memory")
#endif

// WGP-scope prefetch: pulls line into ALL cache levels (ISA 10.5, scope 0).
// (__builtin_prefetch lowers to scope:SCOPE_SE which stops at GL2.)
__device__ __forceinline__ void prefetch_wgp(const void* p) {
    asm volatile("global_prefetch_b8 %0, off" :: "v"(p));
}

__device__ __forceinline__ float sigm_f(float x) {
    return 1.0f / (1.0f + __expf(-x));
}
__device__ __forceinline__ float tanh_f(float x) {
    float e = __expf(2.0f * x);
    return 1.0f - 2.0f / (e + 1.0f);
}

// A-fragment (16x32 bf16) from an LDS row, contiguous k order, matched by B:
// lanes 0-15: K = k0+{0..15}; lanes 16-31: K = k0+16+{0..15}.
__device__ __forceinline__ v16bf lds_a_frag(const __bf16* row, int k0, int kh) {
    union { v16bf v; v8bf h[2]; } u;
    u.h[0] = *(const v8bf*)(row + k0 + kh * 16);
    u.h[1] = *(const v8bf*)(row + k0 + kh * 16 + 8);
    return u.v;
}

// One GEMM pass: 8 accumulators = gates {i,f,g,o} x hidden-col tiles {jA,jB}.
// B addresses: scalar per-tile base + one shared lane-varying offset (voff*).
// Each k-iteration: load 8 independent B fragments, then 8 WMMAs.
__device__ __forceinline__ void lstm_gemm_pass(
    const __bf16* __restrict__ Wih, const __bf16* __restrict__ Whh,
    const __bf16* xrow, const __bf16* hrow,
    int jA, int jB, int voffX, int voffH, int kh, v8f acc[8])
{
    const v8f vzero = {};

    // keep the peel's loads from being hoisted above the preceding barrier
    // (otherwise they live across the elementwise region and get spilled)
    SCHED_FENCE();

    // ---- peeled first x k-step: C = literal zero (folds to inline 0)
    {
        v16bf a = lds_a_frag(xrow, 0, kh);
        v16bf b[8];
#pragma unroll
        for (int i = 0; i < 8; i++) {
            const int colbase = (i & 3) * HID + ((i < 4) ? jA : jB);   // scalar
            b[i] = *(const v16bf*)(Wih + (size_t)colbase * IND + voffX);
        }
#pragma unroll
        for (int i = 0; i < 8; i++)
            acc[i] = __builtin_amdgcn_wmma_f32_16x16x32_bf16(
                false, a, false, b[i], (short)0, vzero, false, false);
        SCHED_FENCE();
    }

    // ---- remaining x contribution, K = 32..128
#pragma clang loop unroll(disable)
    for (int k0 = 32; k0 < IND; k0 += 32) {
        v16bf a = lds_a_frag(xrow, k0, kh);
        v16bf b[8];
#pragma unroll
        for (int i = 0; i < 8; i++) {
            const int colbase = (i & 3) * HID + ((i < 4) ? jA : jB);   // scalar
            const __bf16* bp = Wih + (size_t)colbase * IND + voffX + k0;
            b[i] = *(const v16bf*)bp;
            prefetch_wgp(bp + 32);
        }
#pragma unroll
        for (int i = 0; i < 8; i++)
            acc[i] = __builtin_amdgcn_wmma_f32_16x16x32_bf16(
                false, a, false, b[i], (short)0, acc[i], false, false);
        SCHED_FENCE();
    }

    // ---- h contribution, K = 1024
#pragma clang loop unroll(disable)
    for (int k0 = 0; k0 < HID; k0 += 32) {
        v16bf a = lds_a_frag(hrow, k0, kh);
        v16bf b[8];
#pragma unroll
        for (int i = 0; i < 8; i++) {
            const int colbase = (i & 3) * HID + ((i < 4) ? jA : jB);   // scalar
            const __bf16* bp = Whh + (size_t)colbase * HID + voffH + k0;
            b[i] = *(const v16bf*)bp;
            // warm next k-step into all cache levels: zero register cost
            prefetch_wgp(bp + 32);
        }
#pragma unroll
        for (int i = 0; i < 8; i++)
            acc[i] = __builtin_amdgcn_wmma_f32_16x16x32_bf16(
                false, a, false, b[i], (short)0, acc[i], false, false);
        SCHED_FENCE();
    }
}

// Gate activations + cell-state update for one 16-col tile.
// c lives in a per-thread LDS slot; returns new h packed to bf16 (4 VGPRs).
__device__ __forceinline__ v8bf lstm_update(const v8f* acc4, const float* bsum,
                                            int j, v8f* cslot)
{
    const float bi = bsum[0 * HID + j];
    const float bf = bsum[1 * HID + j];
    const float bg = bsum[2 * HID + j];
    const float bo = bsum[3 * HID + j];
    v8f c = *cslot;
    v8bf h;
#pragma unroll
    for (int r = 0; r < 8; r++) {
        float ig = sigm_f(acc4[0][r] + bi);
        float fg = sigm_f(acc4[1][r] + bf);
        float gg = tanh_f(acc4[2][r] + bg);
        float og = sigm_f(acc4[3][r] + bo);
        float cc = fg * c[r] + ig * gg;
        c[r] = cc;
        h[r] = (__bf16)(og * tanh_f(cc));
    }
    *cslot = c;
    return h;
}

__global__ __launch_bounds__(NTHR, 1) void seq2seq_lstm_kernel(
    const float*  __restrict__ xin,   // [512,256,128] f32
    const __bf16* __restrict__ WihE,  // [4096,128]  bf16
    const __bf16* __restrict__ WhhE,  // [4096,1024] bf16
    const __bf16* __restrict__ WihD,  // [4096,128]  bf16
    const __bf16* __restrict__ WhhD,  // [4096,1024] bf16
    const __bf16* __restrict__ Fcw,   // [128,1024]  bf16
    const float*  __restrict__ bE,    // [4096] b_ih+b_hh (enc)
    const float*  __restrict__ bD,    // [4096] b_ih+b_hh (dec)
    const float*  __restrict__ fcb,   // [128]
    float*        __restrict__ out)   // [512,128,128] f32
{
    __shared__ __bf16 hbuf[BT][HP];     // ~33 KB
    __shared__ __bf16 xbuf[BT][XP];     // ~4.4 KB
    __shared__ v8f    cbuf[4][NTHR];    // 64 KB: per-thread cell-state slots

    const int tid  = threadIdx.x;
    const int wave = tid >> 5;
    const int lane = tid & 31;
    const int n    = lane & 15;          // B-row / C-column this lane owns
    const int kh   = lane >> 4;          // K-half select
    const int b0   = blockIdx.x * BT;
    const int jw   = wave * 16;          // wave's base hidden column
    const int voffH = n * HID + kh * 16; // shared lane-varying elem offset (K=1024)
    const int voffX = n * IND + kh * 16; // shared lane-varying elem offset (K=128)

    for (int i = tid; i < BT * HP; i += NTHR) ((__bf16*)hbuf)[i] = (__bf16)0.0f;
    {
        const v8f vz = {};
#pragma unroll
        for (int tile = 0; tile < 4; tile++) cbuf[tile][tid] = vz;
    }

    auto cell = [&](const __bf16* __restrict__ Wih,
                    const __bf16* __restrict__ Whh,
                    const float*  __restrict__ bsum) {
        const __bf16* xrow = &xbuf[n][0];
        const __bf16* hrow = &hbuf[n][0];
        v8f acc[8];

        // pass 0: tiles jw, jw+256  (gates consumed immediately -> regs freed)
        lstm_gemm_pass(Wih, Whh, xrow, hrow, jw, jw + 256, voffX, voffH, kh, acc);
        v8bf h0 = lstm_update(&acc[0], bsum, jw + n,       &cbuf[0][tid]);
        v8bf h1 = lstm_update(&acc[4], bsum, jw + 256 + n, &cbuf[1][tid]);

        // pass 1: tiles jw+512, jw+768
        lstm_gemm_pass(Wih, Whh, xrow, hrow, jw + 512, jw + 768, voffX, voffH, kh, acc);

        __syncthreads();   // every wave done READING hbuf/xbuf

        v8bf h2 = lstm_update(&acc[0], bsum, jw + 512 + n, &cbuf[2][tid]);
        v8bf h3 = lstm_update(&acc[4], bsum, jw + 768 + n, &cbuf[3][tid]);

        const int m = kh * 8;
#pragma unroll
        for (int r = 0; r < 8; r++) {
            hbuf[m + r][jw + n]       = h0[r];
            hbuf[m + r][jw + 256 + n] = h1[r];
            hbuf[m + r][jw + 512 + n] = h2[r];
            hbuf[m + r][jw + 768 + n] = h3[r];
        }
        __syncthreads();   // new h visible
    };

    __syncthreads();

    // ---------------- Encoder: 256 steps ----------------
    for (int t = 0; t < TENC; t++) {
        for (int idx = tid; idx < BT * IND; idx += NTHR) {
            int rr = idx >> 7, cc = idx & 127;
            xbuf[rr][cc] = (__bf16)xin[((size_t)(b0 + rr) * TENC + t) * IND + cc];
        }
        __syncthreads();
        cell(WihE, WhhE, bE);
    }

    // ---------------- Decoder: 128 autoregressive steps ----------------
    for (int i = tid; i < BT * XP; i += NTHR) ((__bf16*)xbuf)[i] = (__bf16)0.0f;
    __syncthreads();

    for (int s = 0; s < TDEC; s++) {
        cell(WihD, WhhD, bD);

        // fc projection y = h @ fc_w^T + fc_b  (waves 0..7, 16 cols each)
        if (wave < 8) {
            const int o0 = wave * 16;
            v8f ya = {};
            const __bf16* hrow = &hbuf[n][0];
#pragma clang loop unroll(disable)
            for (int k0 = 0; k0 < HID; k0 += 32) {
                v16bf a = lds_a_frag(hrow, k0, kh);
                v16bf b = *(const v16bf*)(Fcw + (size_t)o0 * HID + voffH + k0);
                ya = __builtin_amdgcn_wmma_f32_16x16x32_bf16(
                    false, a, false, b, (short)0, ya, false, false);
                SCHED_FENCE();
            }
            const float bb = fcb[o0 + n];
#pragma unroll
            for (int r = 0; r < 8; r++) {
                float y = ya[r] + bb;
                int m = kh * 8 + r;
                out[((size_t)(b0 + m) * TDEC + s) * OUTD + (o0 + n)] = y;
                xbuf[m][o0 + n] = (__bf16)y;   // feed back as next input
            }
        }
        __syncthreads();
    }
}

// ---------------- one-shot prep kernels ----------------
__global__ void f32_to_bf16_kernel(const float* __restrict__ s,
                                   __bf16* __restrict__ d, int nelem) {
    int i = blockIdx.x * blockDim.x + threadIdx.x;
    if (i < nelem) d[i] = (__bf16)s[i];
}

__global__ void add_bias_kernel(const float* __restrict__ a,
                                const float* __restrict__ b,
                                float* __restrict__ d, int nelem) {
    int i = blockIdx.x * blockDim.x + threadIdx.x;
    if (i < nelem) d[i] = a[i] + b[i];
}

extern "C" void kernel_launch(void* const* d_in, const int* in_sizes, int n_in,
                              void* d_out, int out_size, void* d_ws, size_t ws_size,
                              hipStream_t stream) {
    (void)in_sizes; (void)n_in; (void)out_size; (void)ws_size;

    const float* xin   = (const float*)d_in[0];   // [512,256,128]
    const float* Wih_e = (const float*)d_in[1];   // [4096,128]
    const float* Whh_e = (const float*)d_in[2];   // [4096,1024]
    const float* bih_e = (const float*)d_in[3];
    const float* bhh_e = (const float*)d_in[4];
    const float* Wih_d = (const float*)d_in[5];   // [4096,128]
    const float* Whh_d = (const float*)d_in[6];   // [4096,1024]
    const float* bih_d = (const float*)d_in[7];
    const float* bhh_d = (const float*)d_in[8];
    const float* fcw   = (const float*)d_in[9];   // [128,1024]
    const float* fcb   = (const float*)d_in[10];  // [128]
    float* out = (float*)d_out;

    // workspace: bf16 weights (~19.2 MB, L2-resident) + summed biases
    __bf16* wih_e = (__bf16*)d_ws;                     // 4096*128
    __bf16* whh_e = wih_e + (size_t)4096 * 128;        // 4096*1024
    __bf16* wih_d = whh_e + (size_t)4096 * 1024;       // 4096*128
    __bf16* whh_d = wih_d + (size_t)4096 * 128;        // 4096*1024
    __bf16* fcw_b = whh_d + (size_t)4096 * 1024;       // 128*1024
    float*  bsum_e = (float*)(fcw_b + (size_t)128 * 1024);
    float*  bsum_d = bsum_e + 4096;

    const int n_ih = 4096 * 128, n_hh = 4096 * 1024, n_fc = 128 * 1024;
    f32_to_bf16_kernel<<<(n_ih + 255) / 256, 256, 0, stream>>>(Wih_e, wih_e, n_ih);
    f32_to_bf16_kernel<<<(n_hh + 255) / 256, 256, 0, stream>>>(Whh_e, whh_e, n_hh);
    f32_to_bf16_kernel<<<(n_ih + 255) / 256, 256, 0, stream>>>(Wih_d, wih_d, n_ih);
    f32_to_bf16_kernel<<<(n_hh + 255) / 256, 256, 0, stream>>>(Whh_d, whh_d, n_hh);
    f32_to_bf16_kernel<<<(n_fc + 255) / 256, 256, 0, stream>>>(fcw,   fcw_b, n_fc);
    add_bias_kernel<<<16, 256, 0, stream>>>(bih_e, bhh_e, bsum_e, 4096);
    add_bias_kernel<<<16, 256, 0, stream>>>(bih_d, bhh_d, bsum_d, 4096);

    // 32 workgroups (one 16-row batch tile each), 512 threads = 16 waves
    seq2seq_lstm_kernel<<<dim3(BATCH / BT), dim3(NTHR), 0, stream>>>(
        xin, wih_e, whh_e, wih_d, whh_d, fcw_b, bsum_e, bsum_d, fcb, out);
}
